// StackRNN_50989851738129
// MI455X (gfx1250) — matrix operand order repaired and checked by reading, not compiled, not run on hardware
//
#include <hip/hip_runtime.h>

// ---------------- problem constants ----------------
#define VOCAB    128
#define HID      256
#define NSTATES  64
#define SDEPTH   64
#define SVOCAB   16
#define NACT     18
#define BATCHSZ  256
#define SEQ      1024
// fused/padded layout
#define NCOL     224   // [0,18) mem-actions | [18,32) pad | [32,160) buf logits | [160,224) state logits
#define KR       96    // recurrent K: 64 (state) + 16 (top) + 16 (zero pad)
#define NTILE    14    // 224 / 16
#define NKC      3     // 96 / 32
#define WGSZ     224   // 7 waves of 32

typedef __attribute__((ext_vector_type(16))) _Float16 v16h;
typedef __attribute__((ext_vector_type(8)))  float    v8f;
typedef __attribute__((__vector_size__(16))) int      v4i_vec;   // matches builtin prototype pointee

#if __has_builtin(__builtin_amdgcn_global_load_async_to_lds_b128) && \
    __has_builtin(__builtin_amdgcn_s_wait_asynccnt)
#define HAVE_ASYNC_LDS 1
#else
#define HAVE_ASYNC_LDS 0
#endif

#if HAVE_ASYNC_LDS
// Low 32 bits of a generic shared pointer ARE the LDS byte address (flat aperture rule);
// generic global address == AS1 address. Integer casts avoid cross-AS pointer casts.
__device__ __forceinline__ void async_copy16(const void* gsrc, void* ldst) {
  __builtin_amdgcn_global_load_async_to_lds_b128(
      (__attribute__((address_space(1))) v4i_vec*)(unsigned long long)gsrc,
      (__attribute__((address_space(3))) v4i_vec*)(unsigned int)(unsigned long long)ldst,
      0, 0);
}
#endif

__device__ __forceinline__ float4 f4scale(float a, const float4& x) {
  return make_float4(a * x.x, a * x.y, a * x.z, a * x.w);
}
__device__ __forceinline__ float4 f4axpy(float a, const float4& x, const float4& y) {
  return make_float4(fmaf(a, x.x, y.x), fmaf(a, x.y, y.y), fmaf(a, x.z, y.z), fmaf(a, x.w, y.w));
}
__device__ __forceinline__ float4 f4add(const float4& x, const float4& y) {
  return make_float4(x.x + y.x, x.y + y.y, x.z + y.z, x.w + y.w);
}

// -------- fused-weight column mapping: W_all = [w_mem | pad | w_buf | w_st] --------
__device__ __forceinline__ float wcol(const float* w_mem, const float* w_buf, const float* w_st,
                                      int row, int n) {
  if (n < NACT) return w_mem[row * NACT + n];
  if (n < 32)   return 0.f;
  if (n < 160)  return w_buf[row * VOCAB + (n - 32)];
  return w_st[row * NSTATES + (n - 160)];
}
__device__ __forceinline__ float bcol(const float* b_mem, const float* b_buf, const float* b_st, int n) {
  if (n < NACT) return b_mem[n];
  if (n < 32)   return 0.f;
  if (n < 160)  return b_buf[n - 32];
  return b_st[n - 160];
}

// ---------------- P1: token table E (128x224, all biases folded) and fused W_R (96x224) ----------------
__global__ void precompute_tables(const float* embed, const float* w_state, const float* b_state,
                                  const float* w_top, const float* b_top,
                                  const float* w_mem, const float* b_mem,
                                  const float* w_buf, const float* b_buf,
                                  const float* w_st,  const float* b_st,
                                  float* E, float* WR) {
  int id = blockIdx.x * blockDim.x + threadIdx.x;
  const int nE = VOCAB * NCOL;  // 28672
  const int nW = KR * NCOL;     // 21504
  if (id < nE) {
    int tok = id / NCOL, n = id % NCOL;
    float acc = bcol(b_mem, b_buf, b_st, n);
    for (int j = 0; j < HID; ++j) acc += b_state[j] * wcol(w_mem, w_buf, w_st, HID + j, n);
    for (int j = 0; j < HID; ++j) acc += b_top[j]   * wcol(w_mem, w_buf, w_st, 2 * HID + j, n);
    for (int k = 0; k < HID; ++k) acc += embed[tok * HID + k] * wcol(w_mem, w_buf, w_st, k, n);
    E[id] = acc;
  } else if (id < nE + nW) {
    int r = id - nE;
    int k = r / NCOL, n = r % NCOL;
    float acc = 0.f;
    if (k < NSTATES) {
      for (int j = 0; j < HID; ++j) acc += w_state[k * HID + j] * wcol(w_mem, w_buf, w_st, HID + j, n);
    } else if (k < NSTATES + SVOCAB) {
      int kk = k - NSTATES;
      for (int j = 0; j < HID; ++j) acc += w_top[kk * HID + j] * wcol(w_mem, w_buf, w_st, 2 * HID + j, n);
    }
    WR[r] = acc;  // rows 80..95 stay zero
  }
}

// ---------------- P2: swizzle W_R into f16 WMMA B-operand fragments ----------------
__global__ void swizzle_wr(const float* WR, unsigned int* blob) {
  int id = blockIdx.x * blockDim.x + threadIdx.x;  // 42*32*8 = 10752 dwords
  if (id >= NTILE * NKC * 32 * 8) return;
  int j = id & 7, lane = (id >> 3) & 31, f = id >> 8;
  int tile = f / NKC, kc = f % NKC;
  int N = tile * 16 + (lane & 15);
  int K = kc * 32 + ((lane >> 4) << 4) + 2 * j;  // B k order used here (contiguous, not ISA table order): lanes 0-15 carry K 0..15, lanes 16-31 K 16..31
  union { _Float16 h[2]; unsigned int u; } cv;
  cv.h[0] = (_Float16)WR[K * NCOL + N];
  cv.h[1] = (_Float16)WR[(K + 1) * NCOL + N];
  blob[id] = cv.u;
}

// ---------------- LDS layout (dynamic shared, bytes) ----------------
#define OFF_E      0         // 128*224*4 = 114688
#define OFF_BLOB   114688    // 10752*4   = 43008
#define OFF_STK0   157696    // 16*64*16*4 = 65536
#define OFF_STK1   223232    // 65536
#define OFF_ACC    288768    // 16*224*4  = 14336
#define OFF_RB     303104    // 16*96*2   = 3072 (f16)
#define OFF_PROBS  306176    // 16*32*4   = 2048  (row: push[0..15], pop@16, noop@17, psum@18)
#define OFF_STATE  308224    // 16*64*4   = 4096
#define OFF_TOK    312320    // 16*4
#define LDS_TOTAL  312384

// ---------------- main persistent scan kernel: one 16-row batch tile per workgroup ----------------
__global__ __launch_bounds__(WGSZ, 1)
void stackrnn_scan(const int* __restrict__ x, const float* __restrict__ Eg,
                   const unsigned int* __restrict__ blobg,
                   float* __restrict__ out_logits, float* __restrict__ out_stack,
                   float* __restrict__ out_state) {
  extern __shared__ char smem[];
  float*        E_lds    = (float*)(smem + OFF_E);
  unsigned int* blob_lds = (unsigned int*)(smem + OFF_BLOB);
  float*        stackA   = (float*)(smem + OFF_STK0);
  float*        stackB   = (float*)(smem + OFF_STK1);
  float*        acc      = (float*)(smem + OFF_ACC);
  _Float16*     Rb       = (_Float16*)(smem + OFF_RB);
  float*        probs    = (float*)(smem + OFF_PROBS);
  float*        state_f  = (float*)(smem + OFF_STATE);
  int*          toks     = (int*)(smem + OFF_TOK);

  const int tid  = threadIdx.x;
  const int lane = tid & 31;
  const int wave = tid >> 5;        // 0..6, owns N-tiles 2w, 2w+1
  const int b0   = blockIdx.x * 16;
  const int rA   = lane & 15;
  const int hsel = lane >> 4;       // 0/1: lane half

  // ---- one-time staging into LDS (ASYNCcnt DMA path when available) ----
#if HAVE_ASYNC_LDS
  for (int i = tid * 4; i < VOCAB * NCOL; i += WGSZ * 4)
    async_copy16(Eg + i, E_lds + i);
  for (int i = tid * 4; i < NTILE * NKC * 256; i += WGSZ * 4)
    async_copy16(blobg + i, blob_lds + i);
#else
  for (int i = tid; i < VOCAB * NCOL;      i += WGSZ) E_lds[i]    = Eg[i];
  for (int i = tid; i < NTILE * NKC * 256; i += WGSZ) blob_lds[i] = blobg[i];
#endif
  for (int i = tid; i < 16 * SDEPTH * SVOCAB; i += WGSZ) stackA[i] = ((i & 15) == 0) ? 1.f : 0.f;
  for (int i = tid; i < 16 * KR;              i += WGSZ) Rb[i]     = (_Float16)0.f;
#if HAVE_ASYNC_LDS
  __builtin_amdgcn_s_wait_asynccnt(0);
#endif
  __syncthreads();
  if (tid < 16) Rb[tid * KR + 64] = (_Float16)1.f;   // initial top = one_hot(STACK_NULL)

  // ---- loop-invariant B fragments live in VGPRs for the whole scan ----
  v16h bfrag[2][NKC];
  #pragma unroll
  for (int tt = 0; tt < 2; ++tt) {
    int tile = wave * 2 + tt;
    #pragma unroll
    for (int kc = 0; kc < NKC; ++kc) {
      const int4* p = (const int4*)(blob_lds + ((tile * NKC + kc) * 32 + lane) * 8);
      union { int4 q[2]; v16h h; } u;
      u.q[0] = p[0]; u.q[1] = p[1];
      bfrag[tt][kc] = u.h;
    }
  }
  __syncthreads();

  float* scur = stackA;
  float* snxt = stackB;
  const int tile0 = wave * 2, tile1 = wave * 2 + 1;

  for (int t = 0; t < SEQ; ++t) {
    if (tid < 16) toks[tid] = x[(b0 + tid) * SEQ + t];
    __syncthreads();

    // ---- phase 1: acc[16x224] = E[tok] + R(16x96,f16) x W_R(96x224,f16), f32 accumulate ----
    {
      const int* Rb32 = (const int*)Rb;   // f16 pairs
      v16h afrag[NKC];
      #pragma unroll
      for (int kc = 0; kc < NKC; ++kc) {
        // A k order used here (contiguous, not ISA table order): lanes 0-15 carry K 0..15, lanes 16-31 K 16..31 (matches B blob)
        const int4* p0 = (const int4*)(Rb32 + rA * 48 + kc * 16 + hsel * 8);
        const int4* p1 = (const int4*)(Rb32 + rA * 48 + kc * 16 + 4 + hsel * 8);
        union { int4 q[2]; v16h h; } u;
        u.q[0] = *p0; u.q[1] = *p1;
        afrag[kc] = u.h;
      }
      v8f c0, c1;
      #pragma unroll
      for (int v = 0; v < 8; ++v) {
        int M = v + 8 * hsel;   // C/D layout: VGPR v, lane half selects M or M+8
        c0[v] = E_lds[toks[M] * NCOL + tile0 * 16 + rA];
        c1[v] = E_lds[toks[M] * NCOL + tile1 * 16 + rA];
      }
      // interleave the two independent accumulator chains to hide WMMA RAW latency
      #pragma unroll
      for (int kc = 0; kc < NKC; ++kc) {
        c0 = __builtin_amdgcn_wmma_f32_16x16x32_f16(false, afrag[kc], false, bfrag[0][kc],
                                                    (short)0, c0, false, false);
        c1 = __builtin_amdgcn_wmma_f32_16x16x32_f16(false, afrag[kc], false, bfrag[1][kc],
                                                    (short)0, c1, false, false);
      }
      #pragma unroll
      for (int v = 0; v < 8; ++v) {
        acc[(v + 8 * hsel) * NCOL + tile0 * 16 + rA] = c0[v];
        acc[(v + 8 * hsel) * NCOL + tile1 * 16 + rA] = c1[v];
      }
    }
    __syncthreads();

    // ---- phase 2a: softmaxes + logits writeout ----
    if (tid < 16) {                       // action softmax over 18 cols
      int r = tid;
      const float* a = acc + r * NCOL;
      float m = a[0];
      for (int j = 1; j < NACT; ++j) m = fmaxf(m, a[j]);
      float p[NACT]; float s = 0.f;
      for (int j = 0; j < NACT; ++j) { p[j] = __expf(a[j] - m); s += p[j]; }
      float inv = 1.f / s, ppush = 0.f;
      for (int j = 0; j < NACT; ++j) p[j] *= inv;
      for (int j = 2; j < NACT; ++j) { probs[r * 32 + (j - 2)] = p[j]; ppush += p[j]; }
      probs[r * 32 + 16] = p[0];          // pop
      probs[r * 32 + 17] = p[1];          // noop
      probs[r * 32 + 18] = ppush;         // total push
    } else if (tid < 32) {                // state softmax over 64 cols -> next state (f32 + f16 into R)
      int r = tid - 16;
      const float* a = acc + r * NCOL + 160;
      float m = a[0];
      for (int j = 1; j < NSTATES; ++j) m = fmaxf(m, a[j]);
      float s = 0.f;
      for (int j = 0; j < NSTATES; ++j) { float e = __expf(a[j] - m); state_f[r * NSTATES + j] = e; s += e; }
      float inv = 1.f / s;
      unsigned int* Rbi = (unsigned int*)Rb;
      for (int j = 0; j < NSTATES; j += 2) {
        float p0 = state_f[r * NSTATES + j] * inv;
        float p1 = state_f[r * NSTATES + j + 1] * inv;
        state_f[r * NSTATES + j] = p0;
        state_f[r * NSTATES + j + 1] = p1;
        union { _Float16 h[2]; unsigned int u; } pk;
        pk.h[0] = (_Float16)p0; pk.h[1] = (_Float16)p1;
        Rbi[r * 48 + (j >> 1)] = pk.u;
      }
    } else {                              // stream logits tile (cols 32..159) to HBM, 16B vectors
      for (int e = tid - 32; e < 16 * (VOCAB / 4); e += WGSZ - 32) {
        int r = e >> 5, cq = e & 31;
        float4 vv = *(const float4*)(acc + r * NCOL + 32 + 4 * cq);
        *(float4*)(out_logits + ((size_t)(b0 + r) * SEQ + t) * VOCAB + 4 * cq) = vv;
      }
    }
    __syncthreads();

    // ---- phase 2b: soft stack update (ping-pong, float4 over vocab), new top -> R ----
    for (int e = tid; e < 16 * SDEPTH * (SVOCAB / 4); e += WGSZ) {
      int r = e >> 8, rem = e & 255, d = rem >> 2, vg = rem & 3;
      const float4* srow = (const float4*)(scur + r * (SDEPTH * SVOCAB));
      float4 cur = srow[d * 4 + vg];
      float4 dn  = d ? srow[(d - 1) * 4 + vg] : make_float4(0.f, 0.f, 0.f, 0.f);
      float4 up  = (d < SDEPTH - 1) ? srow[(d + 1) * 4 + vg]
                                    : (vg == 0 ? make_float4(1.f, 0.f, 0.f, 0.f)
                                               : make_float4(0.f, 0.f, 0.f, 0.f));
      float ppush = probs[r * 32 + 18], ppop = probs[r * 32 + 16], pnoop = probs[r * 32 + 17];
      float4 nv = f4axpy(ppush, dn, f4axpy(ppop, up, f4scale(pnoop, cur)));
      if (d == 0) {
        nv = f4add(nv, *(const float4*)(probs + r * 32 + 4 * vg));   // += push_probs
        union { _Float16 h[4]; uint2 u2; } pk;
        pk.h[0] = (_Float16)nv.x; pk.h[1] = (_Float16)nv.y;
        pk.h[2] = (_Float16)nv.z; pk.h[3] = (_Float16)nv.w;
        *(uint2*)((unsigned int*)Rb + r * 48 + 32 + 2 * vg) = pk.u2;  // next top, packed f16
      }
      ((float4*)(snxt + r * (SDEPTH * SVOCAB)))[d * 4 + vg] = nv;
    }
    { float* tsw = scur; scur = snxt; snxt = tsw; }
    // loop-top barrier orders Rb/stack for the next iteration
  }
  __syncthreads();

  // ---- final carry outputs ----
  {
    float4*       dst = (float4*)(out_stack + (size_t)b0 * SDEPTH * SVOCAB);
    const float4* src = (const float4*)scur;
    for (int i = tid; i < 16 * SDEPTH * SVOCAB / 4; i += WGSZ) dst[i] = src[i];
  }
  for (int i = tid; i < 16 * NSTATES; i += WGSZ)
    out_state[(size_t)b0 * NSTATES + i] = state_f[i];
}

// ---------------- host launcher ----------------
extern "C" void kernel_launch(void* const* d_in, const int* in_sizes, int n_in,
                              void* d_out, int out_size, void* d_ws, size_t ws_size,
                              hipStream_t stream) {
  (void)in_sizes; (void)n_in; (void)out_size; (void)ws_size;
  const int*   x       = (const int*)d_in[0];
  const float* embed   = (const float*)d_in[1];
  const float* w_state = (const float*)d_in[2];
  const float* b_state = (const float*)d_in[3];
  const float* w_top   = (const float*)d_in[4];
  const float* b_top   = (const float*)d_in[5];
  const float* w_mem   = (const float*)d_in[6];
  const float* b_mem   = (const float*)d_in[7];
  const float* w_buf   = (const float*)d_in[8];
  const float* b_buf   = (const float*)d_in[9];
  const float* w_st    = (const float*)d_in[10];
  const float* b_st    = (const float*)d_in[11];

  float* ws   = (float*)d_ws;
  float* E    = ws;                         // 128*224 f32
  float* WR   = ws + VOCAB * NCOL;          // 96*224 f32
  unsigned int* blob = (unsigned int*)(ws + VOCAB * NCOL + KR * NCOL);  // 10752 dwords

  float* out        = (float*)d_out;
  float* out_logits = out;                                            // 256*1024*128
  float* out_stack  = out + (size_t)BATCHSZ * SEQ * VOCAB;            // 256*64*16
  float* out_state  = out_stack + (size_t)BATCHSZ * SDEPTH * SVOCAB;  // 256*64

  const int nP1 = VOCAB * NCOL + KR * NCOL;      // 50176
  precompute_tables<<<(nP1 + 255) / 256, 256, 0, stream>>>(
      embed, w_state, b_state, w_top, b_top, w_mem, b_mem, w_buf, b_buf, w_st, b_st, E, WR);
  const int nP2 = NTILE * NKC * 32 * 8;          // 10752
  swizzle_wr<<<(nP2 + 255) / 256, 256, 0, stream>>>(WR, blob);

  (void)hipFuncSetAttribute(reinterpret_cast<const void*>(&stackrnn_scan),
                            hipFuncAttributeMaxDynamicSharedMemorySize, LDS_TOTAL);
  stackrnn_scan<<<BATCHSZ / 16, WGSZ, LDS_TOTAL, stream>>>(
      x, E, blob, out_logits, out_stack, out_state);
}
